// TensorProductScoreModel_18476949307846
// MI455X (gfx1250) — matrix operand order repaired and checked by reading, not compiled, not run on hardware
//
#include <hip/hip_runtime.h>
#include <math.h>

typedef __attribute__((ext_vector_type(16))) _Float16 v16h;
typedef __attribute__((ext_vector_type(8)))  float    v8f;

#define NNODES 4096
#define NEDGES 16384
#define HDIM   192
#define CHUNK  4096
#define WNMAX  5376

// ---------------- irreps tables (device constant) ----------------
__constant__ int c_IRN[4] = {1, 3, 5, 6};
__constant__ int c_IRM[4][6] = {
    {32, 0, 0, 0, 0, 0},
    {32, 8, 8, 0, 0, 0},
    {32, 8, 8, 8, 8, 0},
    {32, 8, 8, 8, 8, 32}};
__constant__ int c_IRL[4][6] = {
    {0, 0, 0, 0, 0, 0},
    {0, 1, 2, 0, 0, 0},
    {0, 1, 2, 1, 2, 0},
    {0, 1, 2, 1, 2, 0}};
__constant__ int c_IRP[4][6] = {
    {1, 0, 0, 0, 0, 0},
    {1, -1, 1, 0, 0, 0},
    {1, -1, 1, 1, -1, 0},
    {1, -1, 1, 1, -1, -1}};

// ---------------- small utility kernels ----------------
__global__ void fill_kernel(float* __restrict__ p, float v, int n) {
  int i = blockIdx.x * blockDim.x + threadIdx.x;
  if (i < n) p[i] = v;
}

__global__ void count_kernel(const int* __restrict__ src, float* __restrict__ cnt, int E) {
  int e = blockIdx.x * blockDim.x + threadIdx.x;
  if (e < E) atomicAdd(&cnt[src[e]], 1.0f);
}

__global__ void nsig_kernel(const int* __restrict__ timev, const int* __restrict__ batch,
                            float* __restrict__ nsig, int N) {
  int n = blockIdx.x * blockDim.x + threadIdx.x;
  if (n >= N) return;
  float t = (float)timev[batch[n]];
  const float nl = logf(10000.0f);
#pragma unroll
  for (int j = 0; j < 16; ++j) {
    float f = expf(-nl * (float)j / 16.0f);
    float a = t * f;
    nsig[n * 32 + j]      = sinf(a);
    nsig[n * 32 + 16 + j] = cosf(a);
  }
}

__global__ void nattr_kernel(const float* __restrict__ x, const float* __restrict__ nsig,
                             float* __restrict__ nattr, int N) {
  int i = blockIdx.x * blockDim.x + threadIdx.x;
  if (i >= N * 106) return;
  int n = i / 106, c = i % 106;
  nattr[i] = (c < 74) ? x[n * 74 + c] : nsig[n * 32 + (c - 74)];
}

__global__ void edge_feat_kernel(const float* __restrict__ pos, const float* __restrict__ eain,
                                 const int* __restrict__ src, const int* __restrict__ dst,
                                 const float* __restrict__ nsig,
                                 float* __restrict__ esh, float* __restrict__ eattr, int E) {
  int e = blockIdx.x * blockDim.x + threadIdx.x;
  if (e >= E) return;
  int s = src[e], d = dst[e];
  float vx = pos[d * 3 + 0] - pos[s * 3 + 0];
  float vy = pos[d * 3 + 1] - pos[s * 3 + 1];
  float vz = pos[d * 3 + 2] - pos[s * 3 + 2];
  float dn = sqrtf(vx * vx + vy * vy + vz * vz);
  float inv = 1.0f / (dn + 1e-8f);
  float x = vx * inv, y = vy * inv, z = vz * inv;
  const float s3 = 1.7320508075688772f, s15 = 3.872983346207417f, s5 = 2.23606797749979f;
  float* sh = esh + (size_t)e * 9;
  sh[0] = 1.0f;
  sh[1] = s3 * y; sh[2] = s3 * z; sh[3] = s3 * x;
  sh[4] = s15 * x * y; sh[5] = s15 * y * z;
  sh[6] = 0.5f * s5 * (3.0f * z * z - 1.0f);
  sh[7] = s15 * x * z; sh[8] = 0.5f * s15 * (x * x - y * y);
  float* ea = eattr + (size_t)e * 86;
#pragma unroll
  for (int c = 0; c < 4; ++c) ea[c] = eain[e * 4 + c];
  for (int c = 0; c < 32; ++c) ea[4 + c] = nsig[s * 32 + c];
  const float delta = 5.0f / 49.0f;
  const float coeff = -0.5f / (delta * delta);
  for (int j = 0; j < 50; ++j) {
    float df = dn - (float)j * delta;
    ea[36 + j] = expf(coeff * df * df);
  }
}

__global__ void build_ef_kernel(const float* __restrict__ ea, const float* __restrict__ h,
                                const int* __restrict__ src, const int* __restrict__ dst,
                                float* __restrict__ ef, int E) {
  int i = blockIdx.x * blockDim.x + threadIdx.x;
  if (i >= E * 96) return;
  int e = i / 96, c = i % 96;
  float v;
  if (c < 32)      v = ea[e * 32 + c];
  else if (c < 64) v = h[(size_t)src[e] * HDIM + (c - 32)];
  else             v = h[(size_t)dst[e] * HDIM + (c - 64)];
  ef[i] = v;
}

// ---------------- fragment swizzle (f32 -> fragment-ready f16, zero-padded K) --------
// A fragment layout (ISA 7.12.2, 16-bit A 16x32):
//   lane = l, half h: K = (h&7) + ((h&8)<<1) + ((l>>4)<<3), M = l&15
// out[mt][kt][lane][16 halves] -> per-lane contiguous 32B, per-wave 1KB coalesced.
__global__ void swizzle_a_kernel(const float* __restrict__ A, int lda, int K,
                                 _Float16* __restrict__ out, int KT, int total) {
  int i = blockIdx.x * blockDim.x + threadIdx.x;
  if (i >= total) return;
  int h = i & 15;
  int lane = (i >> 4) & 31;
  int t = i >> 9;
  int kt = t % KT;
  int mt = t / KT;
  int row = mt * 16 + (lane & 15);
  int kA = kt * 32 + (h & 7) + ((h & 8) << 1) + ((lane >> 4) << 3);
  float v = (kA < K) ? A[(size_t)row * lda + kA] : 0.0f;
  out[i] = (_Float16)v;
}

// B fragment layout (matches A): lane l, half h: K = (h&7) + ((h&8)<<1) + ((l>>4)<<3), N = l&15
__global__ void swizzle_b_kernel(const float* __restrict__ W, int K, int O,
                                 _Float16* __restrict__ out, int KT, int total) {
  int i = blockIdx.x * blockDim.x + threadIdx.x;
  if (i >= total) return;
  int h = i & 15;
  int lane = (i >> 4) & 31;
  int t = i >> 9;
  int kt = t % KT;
  int ot = t / KT;
  int col = ot * 16 + (lane & 15);
  int kB = kt * 32 + (h & 7) + ((h & 8) << 1) + ((lane >> 4) << 3);
  float v = (kB < K) ? W[(size_t)kB * O + col] : 0.0f;
  out[i] = (_Float16)v;
}

// ---------------- WMMA GEMM on pre-swizzled fragments ----------------
// Each wave: 32x32 output macro-tile = 4 independent v_wmma_f32_16x16x32_f16 per k-step.
// No bounds checks: M % 128 == 0 (grid exact), O % 32 == 0, K zero-padded at swizzle.
__global__ void wmma_gemm_swz(const _Float16* __restrict__ Aswz,
                              const _Float16* __restrict__ Bswz,
                              const float* __restrict__ bias,
                              float* __restrict__ C, int ldc,
                              int KT, int relu, int mTileBase) {
  const int lane = threadIdx.x & 31;
  const int wave = threadIdx.x >> 5;
  const int mT0 = (blockIdx.y * 4 + wave) * 2;  // local 16-row tile index
  const int oT0 = blockIdx.x * 2;               // 16-col tile index
  const size_t tileStride = (size_t)KT * 512;   // halves per (tile, all k)
  const _Float16* Ap0 = Aswz + (size_t)(mTileBase + mT0) * tileStride + (size_t)lane * 16;
  const _Float16* Ap1 = Ap0 + tileStride;
  const _Float16* Bp0 = Bswz + (size_t)oT0 * tileStride + (size_t)lane * 16;
  const _Float16* Bp1 = Bp0 + tileStride;
  v8f acc00 = {}, acc01 = {}, acc10 = {}, acc11 = {};
  for (int kt = 0; kt < KT; ++kt) {
    __builtin_prefetch(Bp0 + 512, 0, 1);
    __builtin_prefetch(Bp1 + 512, 0, 1);
    v16h a0 = *(const v16h*)Ap0;
    v16h a1 = *(const v16h*)Ap1;
    v16h b0 = *(const v16h*)Bp0;
    v16h b1 = *(const v16h*)Bp1;
    Ap0 += 512; Ap1 += 512; Bp0 += 512; Bp1 += 512;
    acc00 = __builtin_amdgcn_wmma_f32_16x16x32_f16(false, a0, false, b0, (short)0, acc00, false, false);
    acc01 = __builtin_amdgcn_wmma_f32_16x16x32_f16(false, a0, false, b1, (short)0, acc01, false, false);
    acc10 = __builtin_amdgcn_wmma_f32_16x16x32_f16(false, a1, false, b0, (short)0, acc10, false, false);
    acc11 = __builtin_amdgcn_wmma_f32_16x16x32_f16(false, a1, false, b1, (short)0, acc11, false, false);
  }
  const int laneLo = lane & 15;
  const int laneHi = lane >> 4;
  const int col = oT0 * 16 + laneLo;
  const float bc0 = bias[col];
  const float bc1 = bias[col + 16];
#pragma unroll
  for (int r = 0; r < 8; ++r) {
    int m0 = mT0 * 16 + r + (laneHi << 3);
    int m1 = m0 + 16;
    float v00 = acc00[r] + bc0;
    float v01 = acc01[r] + bc1;
    float v10 = acc10[r] + bc0;
    float v11 = acc11[r] + bc1;
    if (relu) {
      v00 = fmaxf(v00, 0.0f); v01 = fmaxf(v01, 0.0f);
      v10 = fmaxf(v10, 0.0f); v11 = fmaxf(v11, 0.0f);
    }
    C[(size_t)m0 * ldc + col]      = v00;
    C[(size_t)m0 * ldc + col + 16] = v01;
    C[(size_t)m1 * ldc + col]      = v10;
    C[(size_t)m1 * ldc + col + 16] = v11;
  }
}

// ---------------- Clebsch-Gordan init (device, double precision) ----------------
__device__ double dfact(int n) {
  double r = 1.0;
  for (int i = 2; i <= n; ++i) r *= (double)i;
  return r;
}

__device__ void umat_fill(int l, double Ur[5][5], double Ui[5][5]) {
  for (int i = 0; i < 5; ++i)
    for (int j = 0; j < 5; ++j) { Ur[i][j] = 0.0; Ui[i][j] = 0.0; }
  Ur[l][l] = 1.0;
  const double is2 = 0.70710678118654752440;
  for (int m = 1; m <= l; ++m) {
    double sgn = (m & 1) ? -1.0 : 1.0;
    Ur[l + m][l + m] = sgn * is2;
    Ur[l + m][l - m] = is2;
    Ui[l - m][l + m] = -sgn * is2;
    Ui[l - m][l - m] = is2;
  }
}

__global__ void cg_init_kernel(float* __restrict__ cg) {
  int t = blockIdx.x * blockDim.x + threadIdx.x;
  if (t >= 27) return;
  int l1 = t / 9, l2 = (t / 3) % 3, l3 = t % 3;
  float* out = cg + t * 125;
  for (int i = 0; i < 125; ++i) out[i] = 0.0f;
  int lo = (l1 > l2) ? l1 - l2 : l2 - l1;
  if (l3 < lo || l3 > l1 + l2) return;
  int d1 = 2 * l1 + 1, d2 = 2 * l2 + 1, d3 = 2 * l3 + 1;
  double Tc[5][5][5];
  for (int a = 0; a < 5; ++a)
    for (int b = 0; b < 5; ++b)
      for (int c = 0; c < 5; ++c) Tc[a][b][c] = 0.0;
  for (int m1 = -l1; m1 <= l1; ++m1)
    for (int m2 = -l2; m2 <= l2; ++m2) {
      int m3 = m1 + m2;
      if (m3 < -l3 || m3 > l3) continue;
      double pref = sqrt((2.0 * l3 + 1.0) * dfact(l3 + l1 - l2) * dfact(l3 - l1 + l2) *
                         dfact(l1 + l2 - l3) / dfact(l1 + l2 + l3 + 1));
      pref *= sqrt(dfact(l3 + m3) * dfact(l3 - m3) * dfact(l1 - m1) * dfact(l1 + m1) *
                   dfact(l2 - m2) * dfact(l2 + m2));
      double s = 0.0;
      for (int k = 0; k <= l1 + l2 - l3; ++k) {
        int a1 = k, a2 = l1 + l2 - l3 - k, a3 = l1 - m1 - k;
        int a4 = l2 + m2 - k, a5 = l3 - l2 + m1 + k, a6 = l3 - l1 - m2 + k;
        if (a1 < 0 || a2 < 0 || a3 < 0 || a4 < 0 || a5 < 0 || a6 < 0) continue;
        double den = dfact(a1) * dfact(a2) * dfact(a3) * dfact(a4) * dfact(a5) * dfact(a6);
        s += ((k & 1) ? -1.0 : 1.0) / den;
      }
      Tc[l1 + m1][l2 + m2][l3 + m3] = pref * s;
    }
  double U1r[5][5], U1i[5][5], U2r[5][5], U2i[5][5], U3r[5][5], U3i[5][5];
  umat_fill(l1, U1r, U1i);
  umat_fill(l2, U2r, U2i);
  umat_fill(l3, U3r, U3i);
  double Tr[5][5][5], Tg[5][5][5];
  double sumR = 0.0, sumI = 0.0;
  for (int a = 0; a < d1; ++a)
    for (int b = 0; b < d2; ++b)
      for (int c = 0; c < d3; ++c) {
        double sr = 0.0, si = 0.0;
        for (int i = 0; i < d1; ++i)
          for (int j = 0; j < d2; ++j)
            for (int k = 0; k < d3; ++k) {
              double tv = Tc[i][j][k];
              if (tv == 0.0) continue;
              double zr = U1r[a][i] * U2r[b][j] - U1i[a][i] * U2i[b][j];
              double zi = U1r[a][i] * U2i[b][j] + U1i[a][i] * U2r[b][j];
              double wr = zr * U3r[c][k] + zi * U3i[c][k];
              double wi = zi * U3r[c][k] - zr * U3i[c][k];
              sr += wr * tv;
              si += wi * tv;
            }
        Tr[a][b][c] = sr;
        Tg[a][b][c] = si;
        sumR += fabs(sr);
        sumI += fabs(si);
      }
  bool useR = (sumR >= sumI);
  double nrm = 0.0;
  for (int a = 0; a < d1; ++a)
    for (int b = 0; b < d2; ++b)
      for (int c = 0; c < d3; ++c) {
        double v = useR ? Tr[a][b][c] : Tg[a][b][c];
        nrm += v * v;
      }
  nrm = sqrt(nrm);
  if (nrm < 1e-30) return;
  for (int a = 0; a < d1; ++a)
    for (int b = 0; b < d2; ++b)
      for (int c = 0; c < d3; ++c)
        out[a * 25 + b * 5 + c] = (float)((useR ? Tr[a][b][c] : Tg[a][b][c]) / nrm);
}

// ---------------- tensor product + segment scatter ----------------
__global__ void tp_kernel(const float* __restrict__ h, const float* __restrict__ esh,
                          const float* __restrict__ wbuf, float* __restrict__ sacc,
                          const int* __restrict__ src, const int* __restrict__ dst,
                          int e0, int ecount, int seqIn, int seqOut, int WN,
                          const float* __restrict__ cg) {
  int t = blockIdx.x * blockDim.x + threadIdx.x;
  if (t >= ecount) return;
  int e = e0 + t;
  const int nIn = c_IRN[seqIn], nOut = c_IRN[seqOut];
  int inOff[6], outOff[6], nin[6];
  {
    int o = 0;
    for (int i = 0; i < nIn; ++i) {
      inOff[i] = o;
      o += c_IRM[seqIn][i] * (2 * c_IRL[seqIn][i] + 1);
    }
  }
  {
    int o = 0;
    for (int i = 0; i < nOut; ++i) {
      outOff[i] = o;
      nin[i] = 0;
      o += c_IRM[seqOut][i] * (2 * c_IRL[seqOut][i] + 1);
    }
  }
  const int shL[3] = {0, 1, 2};
  const int shP[3] = {1, -1, 1};
  const int shO[3] = {0, 1, 4};
  for (int i1 = 0; i1 < nIn; ++i1) {
    int l1 = c_IRL[seqIn][i1], p1 = c_IRP[seqIn][i1];
    for (int i2 = 0; i2 < 3; ++i2) {
      int l2 = shL[i2], p2 = shP[i2];
      for (int i3 = 0; i3 < nOut; ++i3) {
        int l3 = c_IRL[seqOut][i3];
        if (c_IRP[seqOut][i3] != p1 * p2) continue;
        int lo = (l1 > l2) ? l1 - l2 : l2 - l1;
        if (l3 < lo || l3 > l1 + l2) continue;
        nin[i3]++;
      }
    }
  }
  const float* x  = h + (size_t)dst[e] * HDIM;
  const float* sh = esh + (size_t)e * 9;
  const float* wr = wbuf + (size_t)t * WN;
  float* so = sacc + (size_t)src[e] * HDIM;
  int w0 = 0;
  for (int i1 = 0; i1 < nIn; ++i1) {
    int m1 = c_IRM[seqIn][i1], l1 = c_IRL[seqIn][i1], p1 = c_IRP[seqIn][i1];
    for (int i2 = 0; i2 < 3; ++i2) {
      int l2 = shL[i2], p2 = shP[i2];
      for (int i3 = 0; i3 < nOut; ++i3) {
        int m3 = c_IRM[seqOut][i3], l3 = c_IRL[seqOut][i3];
        if (c_IRP[seqOut][i3] != p1 * p2) continue;
        int lo = (l1 > l2) ? l1 - l2 : l2 - l1;
        if (l3 < lo || l3 > l1 + l2) continue;
        int d1 = 2 * l1 + 1, d2 = 2 * l2 + 1, d3 = 2 * l3 + 1;
        const float* C = cg + ((l1 * 3 + l2) * 3 + l3) * 125;
        float oblk[40];
        for (int q = 0; q < m3 * d3; ++q) oblk[q] = 0.0f;
        for (int u = 0; u < m1; ++u) {
          float tmp[5] = {0.f, 0.f, 0.f, 0.f, 0.f};
          for (int m = 0; m < d1; ++m) {
            float xv = x[inOff[i1] + u * d1 + m];
            for (int n = 0; n < d2; ++n) {
              float xs = xv * sh[shO[i2] + n];
              const float* Cr = C + (m * 25 + n * 5);
              for (int k = 0; k < d3; ++k) tmp[k] += xs * Cr[k];
            }
          }
          const float* Wp = wr + w0 + u * m3;
          for (int wv = 0; wv < m3; ++wv) {
            float wc = Wp[wv];
            for (int k = 0; k < d3; ++k) oblk[wv * d3 + k] += wc * tmp[k];
          }
        }
        float alpha = rsqrtf((float)(m1 * (nin[i3] > 0 ? nin[i3] : 1)));
        for (int wv = 0; wv < m3; ++wv)
          for (int k = 0; k < d3; ++k)
            atomicAdd(&so[outOff[i3] + wv * d3 + k], alpha * oblk[wv * d3 + k]);
        w0 += m1 * m3;
      }
    }
  }
}

__global__ void finalize_kernel(float* __restrict__ hN, const float* __restrict__ hC,
                                const float* __restrict__ cnt, int N, int inDim, int outDim) {
  int i = blockIdx.x * blockDim.x + threadIdx.x;
  if (i >= N * HDIM) return;
  int n = i / HDIM, c = i % HDIM;
  if (c < outDim) {
    float v = hN[i] / fmaxf(cnt[n], 1.0f);
    if (c < inDim) v += hC[i];
    hN[i] = v;
  } else {
    hN[i] = 0.0f;
  }
}

__global__ void bn_stats_kernel(const float* __restrict__ h, float* __restrict__ mean,
                                float* __restrict__ var, int N, int seqOut) {
  int ir = blockIdx.x;
  int m = c_IRM[seqOut][ir], l = c_IRL[seqOut][ir];
  int d = 2 * l + 1;
  int a = 0, ch = 0;
  for (int j = 0; j < ir; ++j) {
    a += c_IRM[seqOut][j] * (2 * c_IRL[seqOut][j] + 1);
    ch += c_IRM[seqOut][j];
  }
  __shared__ float red[256];
  __shared__ float red2[256];
  for (int u = 0; u < m; ++u) {
    float s = 0.0f, s2 = 0.0f;
    for (int idx = threadIdx.x; idx < N * d; idx += blockDim.x) {
      int n = idx / d, k = idx % d;
      float v = h[(size_t)n * HDIM + a + u * d + k];
      s += v;
      s2 += v * v;
    }
    red[threadIdx.x] = s;
    red2[threadIdx.x] = s2;
    __syncthreads();
    for (int st = 128; st > 0; st >>= 1) {
      if ((int)threadIdx.x < st) {
        red[threadIdx.x] += red[threadIdx.x + st];
        red2[threadIdx.x] += red2[threadIdx.x + st];
      }
      __syncthreads();
    }
    if (threadIdx.x == 0) {
      float inv = 1.0f / (float)(N * d);
      float mu = red[0] * inv;
      float mm = (l == 0) ? mu : 0.0f;
      mean[ch + u] = mm;
      var[ch + u] = red2[0] * inv - mm * mm;
    }
    __syncthreads();
  }
}

__global__ void bn_apply_kernel(float* __restrict__ h, const float* __restrict__ mean,
                                const float* __restrict__ var, int N, int seqOut, int outDim) {
  int i = blockIdx.x * blockDim.x + threadIdx.x;
  if (i >= N * outDim) return;
  int n = i / outDim, c = i % outDim;
  int a = 0, ch = 0;
  for (int j = 0; j < c_IRN[seqOut]; ++j) {
    int mm = c_IRM[seqOut][j], d = 2 * c_IRL[seqOut][j] + 1;
    int sz = mm * d;
    if (c < a + sz) {
      ch += (c - a) / d;
      break;
    }
    a += sz;
    ch += mm;
  }
  float v = h[(size_t)n * HDIM + c];
  h[(size_t)n * HDIM + c] = (v - mean[ch]) * rsqrtf(var[ch] + 1e-5f);
}

__global__ void readout_kernel(const float* __restrict__ h, const float* __restrict__ W0,
                               const float* __restrict__ W1, const float* __restrict__ W3,
                               float* __restrict__ out, int N) {
  int n = blockIdx.x * blockDim.x + threadIdx.x;
  if (n >= N) return;
  const float* hn = h + (size_t)n * HDIM;
  const float inv32 = 0.17677669529663687f;
  const float inv8  = 0.35355339059327373f;
  float o0 = 0.f, o1 = 0.f, o2 = 0.f;
  for (int w = 0; w < 32; ++w) {
    float s0 = 0.f;
    for (int c = 0; c < 32; ++c) s0 += hn[c] * W0[c * 96 + 32 + w];
    float g = tanhf(s0 * inv32);
    float m0 = 0.f, m1 = 0.f, m2 = 0.f;
    for (int u = 0; u < 8; ++u) {
      float wu = W1[u * 32 + w];
      m0 += hn[32 + u * 3 + 0] * wu;
      m1 += hn[32 + u * 3 + 1] * wu;
      m2 += hn[32 + u * 3 + 2] * wu;
    }
    float gw = g * W3[w] * inv8;
    o0 += m0 * gw;
    o1 += m1 * gw;
    o2 += m2 * gw;
  }
  out[n * 3 + 0] = o0 * inv32;
  out[n * 3 + 1] = o1 * inv32;
  out[n * 3 + 2] = o2 * inv32;
}

// ---------------- host orchestration ----------------
static inline int ceil_div(int a, int b) { return (a + b - 1) / b; }

extern "C" void kernel_launch(void* const* d_in, const int* in_sizes, int n_in,
                              void* d_out, int out_size, void* d_ws, size_t ws_size,
                              hipStream_t stream) {
  (void)in_sizes; (void)n_in; (void)out_size; (void)ws_size;
  const int N = NNODES, E = NEDGES;

  const float* x        = (const float*)d_in[0];
  const float* pos      = (const float*)d_in[1];
  const float* eattr_in = (const float*)d_in[2];
  const int*   eidx     = (const int*)d_in[3];
  const int*   timev    = (const int*)d_in[4];
  const int*   batch    = (const int*)d_in[5];
  const float* P[28];
  for (int i = 0; i < 28; ++i) P[i] = (const float*)d_in[6 + i];
  const int* srcI = eidx;
  const int* dstI = eidx + E;

  float* ws = (float*)d_ws;
  size_t off = 0;
  auto alloc = [&](size_t nf) { float* p = ws + off; off += (nf + 7) & ~(size_t)7; return p; };
  float* hA    = alloc((size_t)N * HDIM);
  float* hB    = alloc((size_t)N * HDIM);
  float* cnt   = alloc(N);
  float* nsig  = alloc((size_t)N * 32);
  float* nattr = alloc((size_t)N * 106);
  float* t1    = alloc((size_t)N * 32);
  float* eattr = alloc((size_t)E * 86);
  float* t2    = alloc((size_t)E * 32);
  float* ea    = alloc((size_t)E * 32);
  float* esh   = alloc((size_t)E * 9);
  float* ef    = alloc((size_t)E * 96);
  float* hid   = alloc((size_t)E * 96);
  float* wbuf  = alloc((size_t)CHUNK * WNMAX);
  float* cg    = alloc(27 * 125);
  float* meanb = alloc(96);
  float* varb  = alloc(96);
  // fragment-swizzled f16 buffers (stored in float workspace)
  _Float16* Aswz = (_Float16*)alloc((size_t)(E / 16) * 4 * 512 / 2 + 8);  // max M/16 * KT * 512 halves
  _Float16* Bswz = (_Float16*)alloc((size_t)(WNMAX / 16) * 4 * 512 / 2 + 8);

  const int wnArr[4]     = {1536, 2688, 3840, 5376};
  const int inDimArr[4]  = {32, 96, 160, 192};
  const int outDimArr[4] = {96, 160, 192, 192};
  const int seqInArr[4]  = {0, 1, 2, 3};
  const int seqOutArr[4] = {1, 2, 3, 3};
  const int nOutIrArr[4] = {3, 5, 6, 6};

  dim3 b256(256);

  // full GEMM helper: swizzle A and W then one WMMA pass over all rows
  auto gemm = [&](const float* A, int lda, int M, int K, const float* W, const float* bias,
                  float* C, int ldc, int O, int relu) {
    int KT = (K + 31) / 32;
    int aTot = (M / 16) * KT * 512;
    int bTot = (O / 16) * KT * 512;
    swizzle_a_kernel<<<dim3(ceil_div(aTot, 256)), b256, 0, stream>>>(A, lda, K, Aswz, KT, aTot);
    swizzle_b_kernel<<<dim3(ceil_div(bTot, 256)), b256, 0, stream>>>(W, K, O, Bswz, KT, bTot);
    wmma_gemm_swz<<<dim3(O / 32, M / 128), dim3(128), 0, stream>>>(
        Aswz, Bswz, bias, C, ldc, KT, relu, 0);
  };

  cg_init_kernel<<<dim3(1), dim3(32), 0, stream>>>(cg);

  fill_kernel<<<dim3(ceil_div(N, 256)), b256, 0, stream>>>(cnt, 0.0f, N);
  count_kernel<<<dim3(ceil_div(E, 256)), b256, 0, stream>>>(srcI, cnt, E);

  nsig_kernel<<<dim3(ceil_div(N, 256)), b256, 0, stream>>>(timev, batch, nsig, N);
  nattr_kernel<<<dim3(ceil_div(N * 106, 256)), b256, 0, stream>>>(x, nsig, nattr, N);
  fill_kernel<<<dim3(ceil_div(N * HDIM, 256)), b256, 0, stream>>>(hA, 0.0f, N * HDIM);
  gemm(nattr, 106, N, 106, P[0], P[1], t1, 32, 32, 1);   // node MLP layer 1
  gemm(t1, 32, N, 32, P[2], P[3], hA, HDIM, 32, 0);      // node MLP layer 2 -> h[:, :32]

  edge_feat_kernel<<<dim3(ceil_div(E, 256)), b256, 0, stream>>>(pos, eattr_in, srcI, dstI,
                                                                nsig, esh, eattr, E);
  gemm(eattr, 86, E, 86, P[4], P[5], t2, 32, 32, 1);     // edge MLP layer 1
  gemm(t2, 32, E, 32, P[6], P[7], ea, 32, 32, 0);        // edge MLP layer 2

  float* hC = hA;
  float* hN = hB;
  for (int layer = 0; layer < 4; ++layer) {
    const int wn = wnArr[layer];
    const int seqIn = seqInArr[layer], seqOut = seqOutArr[layer];
    const float* fcW1 = P[8 + 4 * layer];
    const float* fcb1 = P[9 + 4 * layer];
    const float* fcW2 = P[10 + 4 * layer];
    const float* fcb2 = P[11 + 4 * layer];

    build_ef_kernel<<<dim3(ceil_div(E * 96, 256)), b256, 0, stream>>>(ea, hC, srcI, dstI, ef, E);
    gemm(ef, 96, E, 96, fcW1, fcb1, hid, 96, 96, 1);     // hid = relu(ef @ W1 + b1)

    // swizzle once for hid (A) and W2 (B); GEMM2 + TP in edge chunks (wbuf capacity)
    {
      int KT = 3;  // K = 96
      int aTot = (E / 16) * KT * 512;
      int bTot = (wn / 16) * KT * 512;
      swizzle_a_kernel<<<dim3(ceil_div(aTot, 256)), b256, 0, stream>>>(hid, 96, 96, Aswz, KT, aTot);
      swizzle_b_kernel<<<dim3(ceil_div(bTot, 256)), b256, 0, stream>>>(fcW2, 96, wn, Bswz, KT, bTot);

      fill_kernel<<<dim3(ceil_div(N * HDIM, 256)), b256, 0, stream>>>(hN, 0.0f, N * HDIM);
      for (int c0 = 0; c0 < E; c0 += CHUNK) {
        int ec = (E - c0 < CHUNK) ? (E - c0) : CHUNK;
        wmma_gemm_swz<<<dim3(wn / 32, ec / 128), dim3(128), 0, stream>>>(
            Aswz, Bswz, fcb2, wbuf, wn, KT, 0, c0 / 16);
        tp_kernel<<<dim3(ceil_div(ec, 256)), b256, 0, stream>>>(
            hC, esh, wbuf, hN, srcI, dstI, c0, ec, seqIn, seqOut, wn, cg);
      }
    }

    finalize_kernel<<<dim3(ceil_div(N * HDIM, 256)), b256, 0, stream>>>(
        hN, hC, cnt, N, inDimArr[layer], outDimArr[layer]);
    bn_stats_kernel<<<dim3(nOutIrArr[layer]), b256, 0, stream>>>(hN, meanb, varb, N, seqOut);
    bn_apply_kernel<<<dim3(ceil_div(N * outDimArr[layer], 256)), b256, 0, stream>>>(
        hN, meanb, varb, N, seqOut, outDimArr[layer]);

    float* tmp = hC; hC = hN; hN = tmp;
  }

  readout_kernel<<<dim3(ceil_div(N, 256)), b256, 0, stream>>>(
      hC, P[24], P[25], P[27], (float*)d_out, N);
}
